// CausalSelfAttention_7791070675065
// MI455X (gfx1250) — matrix operand order repaired and checked by reading, not compiled, not run on hardware
//
#include <hip/hip_runtime.h>
#include <hip/hip_bf16.h>

// ---------------------------------------------------------------------------
// Causal self-attention, B=1, L=4096, C=1024, H=16, D=64 — MI455X (gfx1250)
// GEMMs: v_wmma_f32_16x16x32_bf16; weight tiles staged to LDS by the Tensor
// Data Mover (quarter-tile per wave, double buffered, per-wave
// s_wait_tensorcnt). LDS object escaped into an asm input each stage so AA
// cannot undef-fold TDM-written data while loads keep addrspace(3) (ds_load).
// Stage schedule pinned to "all DS reads, then all WMMAs" via
// sched_group_barrier so the compiler emits partial DScnt waits.
// Attention: flash, K/V register double-buffered.
// ---------------------------------------------------------------------------

typedef __bf16 bf16;
typedef __attribute__((ext_vector_type(16))) __bf16 v16bf;
typedef __attribute__((ext_vector_type(8)))  __bf16 v8bf;
typedef __attribute__((ext_vector_type(8)))  float  v8f;
typedef __attribute__((ext_vector_type(4)))  unsigned v4u;
typedef __attribute__((ext_vector_type(4)))  int      v4i;
typedef __attribute__((ext_vector_type(8)))  int      v8i;

#define SEQ 4096
#define CH  1024
#define NH  16
#define HD  64

// Stage = 64 N-rows x 64 K-cols of Wt; rows padded 128B -> 144B by the TDM
// pad feature (interval 32 DWORDs, amount 4 DWORDs) to spread LDS banks.
#define KB        64
#define ROWPAD    72                   // bf16 per padded row (144 B)
#define STAGE_EL  (64 * ROWPAD)
#define STAGE_BY  (STAGE_EL * 2)
#define NSTAGE    (CH / KB)            // 16

#if defined(__has_builtin)
#if __has_builtin(__builtin_amdgcn_tensor_load_to_lds) && \
    __has_builtin(__builtin_amdgcn_s_wait_tensorcnt)
#define HAVE_TDM 1
#endif
#if __has_builtin(__builtin_amdgcn_sched_group_barrier)
#define HAVE_SGB 1
#endif
#endif

__device__ __forceinline__ v8f wmma_bf16(v16bf a, v16bf b, v8f c) {
  return __builtin_amdgcn_wmma_f32_16x16x32_bf16(
      false, a, false, b, (short)0, c, false, false);
}

// A-operand in this code's contiguous k order (B matches): lane<16 holds K 0..15; lane>=16
// holds K 16..31.
__device__ __forceinline__ v16bf load_a(const bf16* rowbase, int hi) {
  v8bf lo = *(const v8bf*)(rowbase + hi * 16);
  v8bf hh = *(const v8bf*)(rowbase + hi * 16 + 8);
  return __builtin_shufflevector(lo, hh, 0, 1, 2, 3, 4, 5, 6, 7,
                                 8, 9, 10, 11, 12, 13, 14, 15);
}

// B-operand: 16 contiguous bf16 (two 16B-aligned halves).
__device__ __forceinline__ v16bf load_b16(const bf16* p) {
  v8bf lo = *(const v8bf*)(p);
  v8bf hh = *(const v8bf*)(p + 8);
  return __builtin_shufflevector(lo, hh, 0, 1, 2, 3, 4, 5, 6, 7,
                                 8, 9, 10, 11, 12, 13, 14, 15);
}

__device__ __forceinline__ v8f vzero8() {
  v8f z = {0.f, 0.f, 0.f, 0.f, 0.f, 0.f, 0.f, 0.f};
  return z;
}

// ---------------------------------------------------------------------------
__global__ __launch_bounds__(256) void cvt_f32_bf16(const float* __restrict__ in,
                                                    bf16* __restrict__ out, int n) {
  int i = blockIdx.x * 256 + threadIdx.x;
  int stride = gridDim.x * 256;
  for (; i < n; i += stride) out[i] = (bf16)in[i];
}

__global__ __launch_bounds__(256) void transpose_w(const float* __restrict__ in,
                                                   bf16* __restrict__ out) {
  int tid = blockIdx.x * 256 + threadIdx.x;
  if (tid < CH * CH) {
    int k = tid >> 10;
    int n = tid & (CH - 1);
    out[(size_t)n * CH + k] = (bf16)in[(size_t)k * CH + n];
  }
}

// ---------------------------------------------------------------------------
// TDM quarter-stage: 16 rows (grow..grow+15) x 64 cols (k0..) of Wt into LDS
// at byte offset lds_off. D# per CDNA5 ISA 8.3/8.4. Shared array is the only
// LDS object -> base offset 0.
// ---------------------------------------------------------------------------
#if defined(HAVE_TDM)
__device__ __forceinline__ void tdm_stage(const bf16* Bt, int grow, int k0,
                                          unsigned lds_off) {
  unsigned long long ga =
      (unsigned long long)(uintptr_t)(Bt + (size_t)grow * CH + k0);
  v4u g0;
  g0.x = 1u;                                       // count=1, user mode
  g0.y = lds_off;                                  // lds_addr (bytes)
  g0.z = (unsigned)(ga & 0xffffffffull);
  g0.w = (unsigned)((ga >> 32) & 0x1ffffffull) | (2u << 30);  // type=2
  v8i g1;
  g1[0] = (int)0x07110000u;   // data_size=2B, pad_en, interval=32dw, amount=4dw
  g1[1] = (int)(1024u << 16); // tensor_dim0 = 1024 (k extent)
  g1[2] = (int)(1024u << 16); // tensor_dim1 = 1024 (n extent)
  g1[3] = (int)(64u << 16);   // tile_dim0 = 64 (k)
  g1[4] = 16;                 // tile_dim1 = 16 (n rows per wave)
  g1[5] = 1024;               // tensor_dim0_stride
  g1[6] = 0;
  g1[7] = 0;
  v4i z4 = {0, 0, 0, 0};
#if __clang_major__ >= 23
  v8i z8 = {0, 0, 0, 0, 0, 0, 0, 0};
  __builtin_amdgcn_tensor_load_to_lds(g0, g1, z4, z4, z8, 0);
#else
  __builtin_amdgcn_tensor_load_to_lds(g0, g1, z4, z4, 0);
#endif
}
#endif

// Fallback cooperative stage copy (all 128 threads, full 64-row stage).
__device__ __forceinline__ void coop_stage(const bf16* Bt, int n0, int k0,
                                           bf16* dst, int tid) {
#pragma unroll
  for (int s = 0; s < 4; ++s) {
    int idx = tid + s * 128;
    int row = idx >> 3;
    int cs  = (idx & 7) * 8;
    *(v8bf*)(dst + row * ROWPAD + cs) =
        *(const v8bf*)(Bt + (size_t)(n0 + row) * CH + k0 + cs);
  }
}

// ---------------------------------------------------------------------------
// WMMA GEMM: C(4096x1024) = A * Wt^T. Block = 4 waves = 128x64 tile; each
// wave owns 32 rows (two 16-row A tiles sharing every staged B operand).
// mode 0: bf16 (H,L,D) | mode 2: bf16 (H,D,L) | mode 3: f32 (L,C)
// ---------------------------------------------------------------------------
__global__ __launch_bounds__(128) void gemm128x64(const bf16* __restrict__ A,
                                                  const bf16* __restrict__ Bt,
                                                  void* __restrict__ out, int mode) {
  __shared__ bf16 sB[2][STAGE_EL];   // sole LDS object (TDM targets offset 0)

  const int lane = threadIdx.x & 31;
  const int wv   = threadIdx.x >> 5;
  const int tid  = threadIdx.x;
  const int lr   = lane & 15;
  const int hi   = lane >> 4;

  // Opaque guard store: predicate from asm, unprovably false -> the LDS
  // object is "stored-to", so reads of TDM-written sB can't fold to undef.
  {
    int t;
    asm volatile("s_mov_b32 %0, 0" : "=s"(t));
    if (t) sB[0][0] = (bf16)1.0f;
  }

  const int nt   = blockIdx.x & 15;     // 16 N-blocks of 64
  const int mt   = blockIdx.x >> 4;     // 32 M-blocks of 128
  const int n0   = nt * 64;
  const int row0 = mt * 128 + wv * 32;
  const unsigned qoff = (unsigned)(wv * 16 * ROWPAD * 2);   // wave's LDS quarter

  const bf16* Arow0 = A + (size_t)(row0 + lr) * CH;
  const bf16* Arow1 = A + (size_t)(row0 + 16 + lr) * CH;

  // Prologue: two stages in flight (each wave DMAs its own 16-row quarter).
#if defined(HAVE_TDM)
  tdm_stage(Bt, n0 + wv * 16, 0, 0u + qoff);
  tdm_stage(Bt, n0 + wv * 16, KB, (unsigned)STAGE_BY + qoff);
#else
  coop_stage(Bt, n0, 0, sB[0], tid);
  coop_stage(Bt, n0, KB, sB[1], tid);
#endif
  v16bf a00 = load_a(Arow0 + 0, hi), a01 = load_a(Arow0 + 32, hi);
  v16bf a10 = load_a(Arow1 + 0, hi), a11 = load_a(Arow1 + 32, hi);

  v8f c00 = vzero8(), c01 = vzero8(), c02 = vzero8(), c03 = vzero8();
  v8f c10 = vzero8(), c11 = vzero8(), c12 = vzero8(), c13 = vzero8();

#pragma unroll 2
  for (int kb = 0; kb < NSTAGE; ++kb) {
    const int k0 = kb * KB;
    const bool more = (kb + 1 < NSTAGE);

    // Register-prefetch next stage's A while TDM finishes this stage.
    v16bf na00 = a00, na01 = a01, na10 = a10, na11 = a11;
    if (more) {
      na00 = load_a(Arow0 + k0 + KB, hi);
      na01 = load_a(Arow0 + k0 + KB + 32, hi);
      na10 = load_a(Arow1 + k0 + KB, hi);
      na11 = load_a(Arow1 + k0 + KB + 32, hi);
    }
#if defined(HAVE_TDM)
    if (more) __builtin_amdgcn_s_wait_tensorcnt((short)1);
    else      __builtin_amdgcn_s_wait_tensorcnt((short)0);
#endif
    __syncthreads();

    // Escape the LDS pointer as an *input* (definition stays the GEP from
    // the __shared__ array -> loads stay ds_load), with a may-write memory
    // clobber so TDM-written bytes can't be undef-folded and reads stay
    // ordered behind the waits.
    const bf16* sbase = sB[kb & 1];
    asm volatile("" : : "v"(sbase) : "memory");

    const bf16* brow0 = sbase + (size_t)(0 * 16 + lr) * ROWPAD + hi * 16;
    const bf16* brow1 = sbase + (size_t)(1 * 16 + lr) * ROWPAD + hi * 16;
    const bf16* brow2 = sbase + (size_t)(2 * 16 + lr) * ROWPAD + hi * 16;
    const bf16* brow3 = sbase + (size_t)(3 * 16 + lr) * ROWPAD + hi * 16;

    // All 8 B operands (16 ds_load_b128), then 16 WMMAs.
    v16bf b00 = load_b16(brow0),      b10 = load_b16(brow1);
    v16bf b20 = load_b16(brow2),      b30 = load_b16(brow3);
    v16bf b01 = load_b16(brow0 + 32), b11 = load_b16(brow1 + 32);
    v16bf b21 = load_b16(brow2 + 32), b31 = load_b16(brow3 + 32);

    c00 = wmma_bf16(a00, b00, c00);  c10 = wmma_bf16(a10, b00, c10);
    c01 = wmma_bf16(a00, b10, c01);  c11 = wmma_bf16(a10, b10, c11);
    c02 = wmma_bf16(a00, b20, c02);  c12 = wmma_bf16(a10, b20, c12);
    c03 = wmma_bf16(a00, b30, c03);  c13 = wmma_bf16(a10, b30, c13);
    c00 = wmma_bf16(a01, b01, c00);  c10 = wmma_bf16(a11, b01, c10);
    c01 = wmma_bf16(a01, b11, c01);  c11 = wmma_bf16(a11, b11, c11);
    c02 = wmma_bf16(a01, b21, c02);  c12 = wmma_bf16(a11, b21, c12);
    c03 = wmma_bf16(a01, b31, c03);  c13 = wmma_bf16(a11, b31, c13);

#if defined(HAVE_SGB)
    // Pin schedule: issue the 16 DS reads first, then the 16 WMMAs, so the
    // compiler keeps all B operands live and emits partial DScnt waits
    // instead of ds_load -> s_wait_dscnt 0 -> wmma ping-pong.
    __builtin_amdgcn_sched_group_barrier(0x100, 16, 0);  // DS reads
    __builtin_amdgcn_sched_group_barrier(0x008, 16, 0);  // WMMA
#endif

    __syncthreads();   // everyone done reading this buffer before rewrite
#if defined(HAVE_TDM)
    if (kb + 2 < NSTAGE)
      tdm_stage(Bt, n0 + wv * 16, k0 + 2 * KB,
                (unsigned)((kb & 1) * STAGE_BY) + qoff);
#else
    if (kb + 2 < NSTAGE) coop_stage(Bt, n0, k0 + 2 * KB, sB[kb & 1], tid);
#endif
    a00 = na00; a01 = na01; a10 = na10; a11 = na11;
  }

  v8f accs[2][4] = {{c00, c01, c02, c03}, {c10, c11, c12, c13}};
#pragma unroll
  for (int mi = 0; mi < 2; ++mi)
#pragma unroll
    for (int j = 0; j < 4; ++j)
#pragma unroll
      for (int e = 0; e < 8; ++e) {
        const int row = row0 + mi * 16 + e + hi * 8;
        const int col = n0 + j * 16 + lr;
        const float v = accs[mi][j][e];
        if (mode == 3) {
          ((float*)out)[(size_t)row * CH + col] = v;
        } else {
          const int h = col >> 6;
          const int d = col & 63;
          bf16* o = (bf16*)out;
          if (mode == 2)
            o[(size_t)h * HD * SEQ + (size_t)d * SEQ + row] = (bf16)v;  // (H,D,L)
          else
            o[(size_t)h * SEQ * HD + (size_t)row * HD + d] = (bf16)v;   // (H,L,D)
        }
      }
}

// ---------------------------------------------------------------------------
// Flash attention
// ---------------------------------------------------------------------------
__device__ __forceinline__ void load_k_chunk(const bf16* Kh, int k0, int lr, int hi,
                                             v16bf (&bk)[2][2]) {
#pragma unroll
  for (int t = 0; t < 2; ++t) {
    const bf16* krow = Kh + (size_t)(k0 + t * 16 + lr) * HD;
    bk[t][0] = load_b16(krow + 0 + hi * 16);
    bk[t][1] = load_b16(krow + 32 + hi * 16);
  }
}

__device__ __forceinline__ void load_v_chunk(const bf16* Vh, int k0, int lr, int hi,
                                             v16bf (&bv)[4]) {
#pragma unroll
  for (int j = 0; j < 4; ++j)
    bv[j] = load_b16(Vh + (size_t)(j * 16 + lr) * SEQ + k0 + hi * 16);
}

__device__ __forceinline__ void attn_step(
    int k0, int q0, int lr, int hi, v16bf aq0, v16bf aq1,
    const v16bf (&bk)[2][2], const v16bf (&bv)[4],
    float (&m_old)[8], float (&lsum)[8],
    v8f& o0, v8f& o1, v8f& o2, v8f& o3, float (*sPw)[34]) {
  v8f s[2];
#pragma unroll
  for (int t = 0; t < 2; ++t) {
    v8f sv = vzero8();
    sv = wmma_bf16(aq0, bk[t][0], sv);
    sv = wmma_bf16(aq1, bk[t][1], sv);
    s[t] = sv;
  }

  const bool need_mask = (k0 + 31 > q0);
#pragma unroll
  for (int t = 0; t < 2; ++t)
#pragma unroll
    for (int e = 0; e < 8; ++e) {
      float v = s[t][e] * 0.125f;
      if (need_mask) {
        const int kk = k0 + t * 16 + lr;
        const int qq = q0 + e + hi * 8;
        if (kk > qq) v = -3.0e38f;
      }
      s[t][e] = v;
    }

  float f[8];
#pragma unroll
  for (int e = 0; e < 8; ++e) {
    float cm = fmaxf(s[0][e], s[1][e]);
    cm = fmaxf(cm, __shfl_xor(cm, 1, 32));
    cm = fmaxf(cm, __shfl_xor(cm, 2, 32));
    cm = fmaxf(cm, __shfl_xor(cm, 4, 32));
    cm = fmaxf(cm, __shfl_xor(cm, 8, 32));
    const float mn = fmaxf(m_old[e], cm);
    f[e] = __expf(m_old[e] - mn);
    const float p0 = __expf(s[0][e] - mn);
    const float p1 = __expf(s[1][e] - mn);
    s[0][e] = p0; s[1][e] = p1;
    float rs = p0 + p1;
    rs += __shfl_xor(rs, 1, 32);
    rs += __shfl_xor(rs, 2, 32);
    rs += __shfl_xor(rs, 4, 32);
    rs += __shfl_xor(rs, 8, 32);
    lsum[e] = lsum[e] * f[e] + rs;
    m_old[e] = mn;
  }

#pragma unroll
  for (int e = 0; e < 8; ++e) {
    o0[e] *= f[e]; o1[e] *= f[e]; o2[e] *= f[e]; o3[e] *= f[e];
  }

#pragma unroll
  for (int t = 0; t < 2; ++t)
#pragma unroll
    for (int e = 0; e < 8; ++e)
      sPw[e + hi * 8][t * 16 + lr] = s[t][e];
  asm volatile("s_wait_dscnt 0" ::: "memory");

  v16bf pa = {};
#pragma unroll
  for (int j = 0; j < 8; ++j) pa[j] = (bf16)sPw[lr][hi * 16 + j];
#pragma unroll
  for (int j = 0; j < 8; ++j) pa[8 + j] = (bf16)sPw[lr][hi * 16 + 8 + j];

  o0 = wmma_bf16(pa, bv[0], o0);
  o1 = wmma_bf16(pa, bv[1], o1);
  o2 = wmma_bf16(pa, bv[2], o2);
  o3 = wmma_bf16(pa, bv[3], o3);
}

__global__ __launch_bounds__(128) void attn_flash(const bf16* __restrict__ Q,
                                                  const bf16* __restrict__ K,
                                                  const bf16* __restrict__ Vt,
                                                  bf16* __restrict__ Oout) {
  __shared__ float sP[4][16][34];

  const int lane = threadIdx.x & 31;
  const int wv   = threadIdx.x >> 5;
  const int wid  = blockIdx.x * 4 + wv;
  const int qb   = wid & 255;
  const int h    = wid >> 8;
  const int q0   = qb * 16;
  const int lr   = lane & 15;
  const int hi   = lane >> 4;

  const bf16* Qh = Q  + (size_t)h * SEQ * HD;
  const bf16* Kh = K  + (size_t)h * SEQ * HD;
  const bf16* Vh = Vt + (size_t)h * HD * SEQ;

  const bf16* qrow = Qh + (size_t)(q0 + lr) * HD;
  const v16bf aq0 = load_a(qrow + 0, hi);
  const v16bf aq1 = load_a(qrow + 32, hi);

  float m_old[8], lsum[8];
  v8f o0 = vzero8(), o1 = vzero8(), o2 = vzero8(), o3 = vzero8();
#pragma unroll
  for (int e = 0; e < 8; ++e) { m_old[e] = -3.0e38f; lsum[e] = 0.0f; }

  float(*sPw)[34] = sP[wv];
  const int qlast = q0 + 15;

  v16bf bkA[2][2], bvA[4], bkB[2][2], bvB[4];
  load_k_chunk(Kh, 0, lr, hi, bkA);
  load_v_chunk(Vh, 0, lr, hi, bvA);

  int k0 = 0;
  for (;;) {
    {
      const int kn = k0 + 32;
      const bool more = (kn <= qlast);
      if (more) { load_k_chunk(Kh, kn, lr, hi, bkB); load_v_chunk(Vh, kn, lr, hi, bvB); }
      attn_step(k0, q0, lr, hi, aq0, aq1, bkA, bvA, m_old, lsum, o0, o1, o2, o3, sPw);
      if (!more) break;
      k0 = kn;
    }
    {
      const int kn = k0 + 32;
      const bool more = (kn <= qlast);
      if (more) { load_k_chunk(Kh, kn, lr, hi, bkA); load_v_chunk(Vh, kn, lr, hi, bvA); }
      attn_step(k0, q0, lr, hi, aq0, aq1, bkB, bvB, m_old, lsum, o0, o1, o2, o3, sPw);
      if (!more) break;
      k0 = kn;
    }
  }

  float inv[8];
#pragma unroll
  for (int e = 0; e < 8; ++e) inv[e] = 1.0f / lsum[e];

  v8f os[4] = {o0, o1, o2, o3};
#pragma unroll
  for (int j = 0; j < 4; ++j)
#pragma unroll
    for (int e = 0; e < 8; ++e) {
      const int row = q0 + e + hi * 8;
      const int col = h * HD + j * 16 + lr;
      Oout[(size_t)row * CH + col] = (bf16)(os[j][e] * inv[e]);
    }
}

// ---------------------------------------------------------------------------
extern "C" void kernel_launch(void* const* d_in, const int* in_sizes, int n_in,
                              void* d_out, int out_size, void* d_ws, size_t ws_size,
                              hipStream_t stream) {
  (void)in_sizes; (void)n_in; (void)out_size; (void)ws_size;
  const float* x  = (const float*)d_in[0];
  const float* Wq = (const float*)d_in[1];
  const float* Wk = (const float*)d_in[2];
  const float* Wv = (const float*)d_in[3];
  const float* Wo = (const float*)d_in[4];

  char* ws = (char*)d_ws;
  const size_t MB = (size_t)1 << 20;
  bf16* xb    = (bf16*)(ws + 0 * MB);
  bf16* Wqt   = (bf16*)(ws + 8 * MB);
  bf16* Wkt   = (bf16*)(ws + 10 * MB);
  bf16* Wvt   = (bf16*)(ws + 12 * MB);
  bf16* Wot   = (bf16*)(ws + 14 * MB);
  bf16* Qb    = (bf16*)(ws + 16 * MB);  // (H, L, D)
  bf16* Kb    = (bf16*)(ws + 24 * MB);  // (H, L, D)
  bf16* Vtb   = (bf16*)(ws + 32 * MB);  // (H, D, L)
  bf16* attnb = (bf16*)(ws + 40 * MB);  // (L, C)

  cvt_f32_bf16<<<16384, 256, 0, stream>>>(x, xb, SEQ * CH);
  transpose_w<<<4096, 256, 0, stream>>>(Wq, Wqt);
  transpose_w<<<4096, 256, 0, stream>>>(Wk, Wkt);
  transpose_w<<<4096, 256, 0, stream>>>(Wv, Wvt);
  transpose_w<<<4096, 256, 0, stream>>>(Wo, Wot);

  gemm128x64<<<512, 128, 0, stream>>>(xb, Wqt, Qb, 0);
  gemm128x64<<<512, 128, 0, stream>>>(xb, Wkt, Kb, 0);
  gemm128x64<<<512, 128, 0, stream>>>(xb, Wvt, Vtb, 2);

  attn_flash<<<1024, 128, 0, stream>>>(Qb, Kb, Vtb, attnb);

  gemm128x64<<<512, 128, 0, stream>>>(attnb, Wot, d_out, 3);
}
